// MultiHeadedAttention_44066364457292
// MI455X (gfx1250) — matrix operand order repaired and checked by reading, not compiled, not run on hardware
//
#include <hip/hip_runtime.h>

// B=4, S=2048, D=1024, H=16, DH=64
#define Bv  4
#define Sv  2048
#define Dv  1024
#define Hv  16
#define DHv 64

typedef __bf16 bh_t;
typedef __attribute__((ext_vector_type(16))) __bf16 v16bf;
typedef __attribute__((ext_vector_type(8)))  float  v8f;

union FragU { v16bf v; unsigned int u[8]; bh_t e[16]; };

// ---- WMMA operand loaders (layouts per CDNA5 ISA 7.12.2, wave32) ----
// A operand 16x32 bf16: lane m = lane&15, hf = lane>>4.
// VGPR j holds K = (j>>2)*16 + hf*8 + (j&3)*2 , +1  (pair packed in one dword)
__device__ __forceinline__ void load_a_bf16(FragU& f, const bh_t* base, int stride,
                                            int m, int hf, int kbase) {
#pragma unroll
  for (int j = 0; j < 8; ++j) {
    int kk = ((j >> 2) << 4) + (hf << 3) + ((j & 3) << 1);
    f.u[j] = *(const unsigned int*)(base + (size_t)m * stride + kbase + kk);
  }
}
// A operand with on-the-fly f32 -> bf16 conversion
__device__ __forceinline__ void load_a_f32(FragU& f, const float* base, int stride,
                                           int m, int hf, int kbase) {
#pragma unroll
  for (int j = 0; j < 8; ++j) {
    int kk = ((j >> 2) << 4) + (hf << 3) + ((j & 3) << 1);
    const float* p = base + (size_t)m * stride + kbase + kk;
    f.e[2 * j]     = (bh_t)p[0];
    f.e[2 * j + 1] = (bh_t)p[1];
  }
}
// B operand 32x16 bf16, element (k,n) fetched from transposed storage T[n*strideT + k].
// lane n = lane&15, hf = lane>>4.  VGPR j holds K = (j>>2)*16 + hf*8 + (j&3)*2 , +1.
__device__ __forceinline__ void load_b_t(FragU& f, const bh_t* T, int strideT,
                                         int n, int hf, int kbase) {
#pragma unroll
  for (int j = 0; j < 8; ++j) {
    int kk = ((j >> 2) << 4) + (hf << 3) + ((j & 3) << 1);
    f.u[j] = *(const unsigned int*)(T + (size_t)n * strideT + kbase + kk);
  }
}

#define WMMA_BF16(A, Bf, C) \
  __builtin_amdgcn_wmma_f32_16x16x32_bf16(false, (A).v, false, (Bf).v, (short)0, (C), false, false)

// ---------------- weight transpose:  Wt[n*D + k] = bf16(W[k*D + n]) ----------------
__global__ __launch_bounds__(256) void k_wtrans(const float* __restrict__ W,
                                                bh_t* __restrict__ Wt) {
  int idx = blockIdx.x * 256 + threadIdx.x;           // over D*D
  int n = idx >> 10, k = idx & (Dv - 1);
  Wt[idx] = (bh_t)W[(size_t)k * Dv + n];
}

// ---------------- QKV projection GEMM: (B*S x D) @ (D x D) + bias ----------------
// Register-blocked: each wave computes a 16x64 strip (4 accumulators); the A
// fragment is loaded once per k-step and reused across 4 WMMAs (3 b128/WMMA).
// WG = 8 waves -> 128 rows x 64 cols.  Grid = (8192/128) x (1024/64) = 64 x 16.
// vmode 0: out[b][h][s][dh]   vmode 1 (V): out[b][h][dh][s] (transposed for P*V B-operand)
__global__ __launch_bounds__(256) void k_gemm_qkv(const float* __restrict__ X,
                                                  const bh_t* __restrict__ Wt,
                                                  const float* __restrict__ bias,
                                                  bh_t* __restrict__ out,
                                                  int vmode, float scale) {
  int wave = threadIdx.x >> 5, lane = threadIdx.x & 31;
  int lo = lane & 15, hf = lane >> 4;
  int rg = blockIdx.x & 63, ct = blockIdx.x >> 6;     // 64 row-groups(128) x 16 col-groups(64)
  int row0 = rg * 128 + wave * 16;
  int col0 = ct * 64;

  v8f acc0 = {}, acc1 = {}, acc2 = {}, acc3 = {};
  for (int kb = 0; kb < Dv; kb += 32) {
    FragU a;
    load_a_f32(a, X, Dv, row0 + lo, hf, kb);
    FragU b;
    load_b_t(b, Wt, Dv, col0 +  0 + lo, hf, kb); acc0 = WMMA_BF16(a, b, acc0);
    load_b_t(b, Wt, Dv, col0 + 16 + lo, hf, kb); acc1 = WMMA_BF16(a, b, acc1);
    load_b_t(b, Wt, Dv, col0 + 32 + lo, hf, kb); acc2 = WMMA_BF16(a, b, acc2);
    load_b_t(b, Wt, Dv, col0 + 48 + lo, hf, kb); acc3 = WMMA_BF16(a, b, acc3);
  }
#pragma unroll
  for (int ctn = 0; ctn < 4; ++ctn) {
    v8f c = (ctn == 0) ? acc0 : (ctn == 1) ? acc1 : (ctn == 2) ? acc2 : acc3;
    int n = col0 + ctn * 16 + lo;
    float bn = bias[n];
    int hh = n >> 6, dh = n & 63;
#pragma unroll
    for (int r = 0; r < 8; ++r) {
      int m = row0 + r + hf * 8;                      // global row in [0, B*S)
      float v = (c[r] + bn) * scale;
      int bb = m >> 11;                               // / S
      int s  = m & (Sv - 1);
      size_t dst;
      if (vmode == 0) dst = (((size_t)(bb * Hv + hh)) * Sv + s) * DHv + dh;
      else            dst = (((size_t)(bb * Hv + hh)) * DHv + dh) * Sv + s;
      out[dst] = (bh_t)v;
    }
  }
}

// ---------------- output projection GEMM: ctx(bf16) @ Wo + bo -> f32 ----------------
__global__ __launch_bounds__(256) void k_gemm_out(const bh_t* __restrict__ A,
                                                  const bh_t* __restrict__ Wt,
                                                  const float* __restrict__ bias,
                                                  float* __restrict__ out) {
  int wave = threadIdx.x >> 5, lane = threadIdx.x & 31;
  int lo = lane & 15, hf = lane >> 4;
  int rg = blockIdx.x & 63, ct = blockIdx.x >> 6;
  int row0 = rg * 128 + wave * 16;
  int col0 = ct * 64;

  v8f acc0 = {}, acc1 = {}, acc2 = {}, acc3 = {};
  for (int kb = 0; kb < Dv; kb += 32) {
    FragU a;
    load_a_bf16(a, A, Dv, row0 + lo, hf, kb);
    FragU b;
    load_b_t(b, Wt, Dv, col0 +  0 + lo, hf, kb); acc0 = WMMA_BF16(a, b, acc0);
    load_b_t(b, Wt, Dv, col0 + 16 + lo, hf, kb); acc1 = WMMA_BF16(a, b, acc1);
    load_b_t(b, Wt, Dv, col0 + 32 + lo, hf, kb); acc2 = WMMA_BF16(a, b, acc2);
    load_b_t(b, Wt, Dv, col0 + 48 + lo, hf, kb); acc3 = WMMA_BF16(a, b, acc3);
  }
#pragma unroll
  for (int ctn = 0; ctn < 4; ++ctn) {
    v8f c = (ctn == 0) ? acc0 : (ctn == 1) ? acc1 : (ctn == 2) ? acc2 : acc3;
    int n = col0 + ctn * 16 + lo;
    float bn = bias[n];
#pragma unroll
    for (int r = 0; r < 8; ++r) {
      int m = row0 + r + hf * 8;
      out[(size_t)m * Dv + n] = c[r] + bn;
    }
  }
}

// ---------------- attention: one WG per (b, h, 16-row q-tile) ----------------
#define SMEM_P     0
#define SMEM_CTX   (16 * Sv * 2)                 // P: 16x2048 bf16 = 64KB
#define SMEM_RMAX  (SMEM_CTX + 16 * DHv * 4)     // ctx: 16x64 f32 = 4KB
#define SMEM_RSUM  (SMEM_RMAX + 64)
#define SMEM_WMAX  (SMEM_RSUM + 64)
#define SMEM_WSUM  (SMEM_WMAX + 8 * 16 * 4)
#define SMEM_TOTAL (SMEM_WSUM + 8 * 16 * 4)      // ~69.1 KB dynamic LDS

__global__ __launch_bounds__(256) void k_attn(const bh_t* __restrict__ Q,
                                              const bh_t* __restrict__ K,
                                              const bh_t* __restrict__ Vt,
                                              const unsigned char* __restrict__ mask,
                                              bh_t* __restrict__ ctxout,
                                              float* __restrict__ topattn) {
  extern __shared__ char smem[];
  bh_t*  Pl     = (bh_t*) (smem + SMEM_P);
  float* ctxl   = (float*)(smem + SMEM_CTX);
  float* rowmax = (float*)(smem + SMEM_RMAX);
  float* rowsum = (float*)(smem + SMEM_RSUM);
  float* wmax   = (float*)(smem + SMEM_WMAX);
  float* wsum   = (float*)(smem + SMEM_WSUM);

  int tid = threadIdx.x;
  int wave = tid >> 5, lane = tid & 31;
  int lo = lane & 15, hf = lane >> 4;

  int blk = blockIdx.x;
  int qt = blk & 127;              // S/16 = 128 q-tiles
  int h  = (blk >> 7) & 15;
  int b  = blk >> 11;

  const bh_t* Qh = Q  + (((size_t)(b * Hv + h)) * Sv + qt * 16) * DHv;
  const bh_t* Kh = K  + ((size_t)(b * Hv + h)) * Sv * DHv;
  const bh_t* Vh = Vt + ((size_t)(b * Hv + h)) * DHv * Sv;
  const unsigned char* Mh = mask + ((size_t)b * Sv + qt * 16) * Sv;

  for (int i = tid; i < 16 * DHv; i += 256) ctxl[i] = 0.f;

  // Q-tile A operands (K = 0..63 -> two 16x32 fragments), reused across all passes
  FragU qa0, qa1;
  load_a_bf16(qa0, Qh, DHv, lo, hf, 0);
  load_a_bf16(qa1, Qh, DHv, lo, hf, 32);

  // ---- pass A: masked scores -> per-row max (no storage) ----
  float lm[8];
#pragma unroll
  for (int r = 0; r < 8; ++r) lm[r] = -3.0e38f;

  for (int ctile = wave; ctile < Sv / 16; ctile += 8) {
    FragU kb0, kb1;
    load_b_t(kb0, Kh, DHv, ctile * 16 + lo, hf, 0);
    load_b_t(kb1, Kh, DHv, ctile * 16 + lo, hf, 32);
    v8f d = {};
    d = WMMA_BF16(qa0, kb0, d);
    d = WMMA_BF16(qa1, kb1, d);
    int key = ctile * 16 + lo;
#pragma unroll
    for (int r = 0; r < 8; ++r) {
      int m = r + hf * 8;
      float v = d[r];
      if (Mh[(size_t)m * Sv + key]) v = -1.0e18f;
      lm[r] = fmaxf(lm[r], v);
    }
  }
#pragma unroll
  for (int r = 0; r < 8; ++r)
#pragma unroll
    for (int off = 8; off >= 1; off >>= 1)
      lm[r] = fmaxf(lm[r], __shfl_xor(lm[r], off, 32));
  if (lo == 0)
#pragma unroll
    for (int r = 0; r < 8; ++r) wmax[wave * 16 + hf * 8 + r] = lm[r];
  __syncthreads();
  if (tid < 16) {
    float mx = -3.0e38f;
#pragma unroll
    for (int w = 0; w < 8; ++w) mx = fmaxf(mx, wmax[w * 16 + tid]);
    rowmax[tid] = mx;
  }
  __syncthreads();

  // ---- pass B: recompute scores, exp(s - max) -> P (bf16, LDS), row sums ----
  float ls[8];
#pragma unroll
  for (int r = 0; r < 8; ++r) ls[r] = 0.f;

  for (int ctile = wave; ctile < Sv / 16; ctile += 8) {
    FragU kb0, kb1;
    load_b_t(kb0, Kh, DHv, ctile * 16 + lo, hf, 0);
    load_b_t(kb1, Kh, DHv, ctile * 16 + lo, hf, 32);
    v8f d = {};
    d = WMMA_BF16(qa0, kb0, d);
    d = WMMA_BF16(qa1, kb1, d);
    int key = ctile * 16 + lo;
#pragma unroll
    for (int r = 0; r < 8; ++r) {
      int m = r + hf * 8;
      float v = d[r];
      if (Mh[(size_t)m * Sv + key]) v = -1.0e18f;
      float e = __expf(v - rowmax[m]);
      ls[r] += e;
      Pl[(size_t)m * Sv + key] = (bh_t)e;
    }
  }
#pragma unroll
  for (int r = 0; r < 8; ++r)
#pragma unroll
    for (int off = 8; off >= 1; off >>= 1)
      ls[r] += __shfl_xor(ls[r], off, 32);
  if (lo == 0)
#pragma unroll
    for (int r = 0; r < 8; ++r) wsum[wave * 16 + hf * 8 + r] = ls[r];
  __syncthreads();
  if (tid < 16) {
    float sm = 0.f;
#pragma unroll
    for (int w = 0; w < 8; ++w) sm += wsum[w * 16 + tid];
    rowsum[tid] = sm;
  }
  __syncthreads();

  // ---- ctx = P x V : split-K across the 8 waves, ds_add_f32 merge ----
  v8f acc0 = {}, acc1 = {}, acc2 = {}, acc3 = {};
  int k0 = wave * (Sv / 8);
  for (int kb = k0; kb < k0 + Sv / 8; kb += 32) {
    FragU pa;
    load_a_bf16(pa, Pl, Sv, lo, hf, kb);
    FragU vb;
    load_b_t(vb, Vh, Sv, 0 * 16 + lo, hf, kb); acc0 = WMMA_BF16(pa, vb, acc0);
    load_b_t(vb, Vh, Sv, 1 * 16 + lo, hf, kb); acc1 = WMMA_BF16(pa, vb, acc1);
    load_b_t(vb, Vh, Sv, 2 * 16 + lo, hf, kb); acc2 = WMMA_BF16(pa, vb, acc2);
    load_b_t(vb, Vh, Sv, 3 * 16 + lo, hf, kb); acc3 = WMMA_BF16(pa, vb, acc3);
  }
#pragma unroll
  for (int r = 0; r < 8; ++r) {
    int m = r + hf * 8;
    atomicAdd(&ctxl[m * DHv +  0 + lo], acc0[r]);
    atomicAdd(&ctxl[m * DHv + 16 + lo], acc1[r]);
    atomicAdd(&ctxl[m * DHv + 32 + lo], acc2[r]);
    atomicAdd(&ctxl[m * DHv + 48 + lo], acc3[r]);
  }
  __syncthreads();

  // normalize + write ctx (bf16) back to [B*S, D] layout
  for (int i = tid; i < 16 * DHv; i += 256) {
    int m = i >> 6, dh = i & 63;
    size_t row = (size_t)b * Sv + qt * 16 + m;
    ctxout[row * Dv + h * DHv + dh] = (bh_t)(ctxl[i] / rowsum[m]);
  }
  // top_attn = attn[:, 0, :, :]
  if (h == 0) {
    for (int i = tid; i < 16 * Sv; i += 256) {
      int m = i >> 11, kc = i & (Sv - 1);
      topattn[((size_t)b * Sv + qt * 16 + m) * Sv + kc] = (float)Pl[i] / rowsum[m];
    }
  }
}

// ---------------- host launcher ----------------
extern "C" void kernel_launch(void* const* d_in, const int* in_sizes, int n_in,
                              void* d_out, int out_size, void* d_ws, size_t ws_size,
                              hipStream_t stream) {
  // setup_inputs order: key, value, query, mask, Wq, bq, Wk, bk, Wv, bv, Wo, bo
  const float* key   = (const float*)d_in[0];
  const float* value = (const float*)d_in[1];
  const float* query = (const float*)d_in[2];
  const unsigned char* mask = (const unsigned char*)d_in[3];  // jnp bool -> 1 byte/elem
  const float* Wq = (const float*)d_in[4];
  const float* bq = (const float*)d_in[5];
  const float* Wk = (const float*)d_in[6];
  const float* bk = (const float*)d_in[7];
  const float* Wv = (const float*)d_in[8];
  const float* bv = (const float*)d_in[9];
  const float* Wo = (const float*)d_in[10];
  const float* bo = (const float*)d_in[11];

  char* ws = (char*)d_ws;
  const size_t WT_BYTES = (size_t)Dv * Dv * sizeof(bh_t);          // 2 MB each
  const size_t ACT_BYTES = (size_t)Bv * Sv * Dv * sizeof(bh_t);    // 16 MB each
  bh_t* WtQ = (bh_t*)(ws + 0 * WT_BYTES);
  bh_t* WtK = (bh_t*)(ws + 1 * WT_BYTES);
  bh_t* WtV = (bh_t*)(ws + 2 * WT_BYTES);
  bh_t* WtO = (bh_t*)(ws + 3 * WT_BYTES);
  char* act = ws + 4 * WT_BYTES;
  bh_t* Qb  = (bh_t*)(act + 0 * ACT_BYTES);   // [b][h][s][dh]
  bh_t* Kb  = (bh_t*)(act + 1 * ACT_BYTES);   // [b][h][s][dh]
  bh_t* Vtb = (bh_t*)(act + 2 * ACT_BYTES);   // [b][h][dh][s]
  bh_t* Ctx = (bh_t*)(act + 3 * ACT_BYTES);   // [B*S][D]

  float* out = (float*)d_out;
  float* topattn = out + (size_t)Bv * Sv * Dv;

  const int wgrid = (Dv * Dv) / 256;          // 4096
  k_wtrans<<<wgrid, 256, 0, stream>>>(Wq, WtQ);
  k_wtrans<<<wgrid, 256, 0, stream>>>(Wk, WtK);
  k_wtrans<<<wgrid, 256, 0, stream>>>(Wv, WtV);
  k_wtrans<<<wgrid, 256, 0, stream>>>(Wo, WtO);

  const int ggrid = ((Bv * Sv) / 128) * (Dv / 64);  // 64 * 16 = 1024
  k_gemm_qkv<<<ggrid, 256, 0, stream>>>(query, WtQ, bq, Qb, 0, 0.125f);  // 1/sqrt(64)
  k_gemm_qkv<<<ggrid, 256, 0, stream>>>(key,   WtK, bk, Kb, 0, 1.0f);
  k_gemm_qkv<<<ggrid, 256, 0, stream>>>(value, WtV, bv, Vtb, 1, 1.0f);

  k_attn<<<Bv * Hv * (Sv / 16), 256, SMEM_TOTAL, stream>>>(Qb, Kb, Vtb, mask, Ctx, topattn);

  k_gemm_out<<<ggrid, 256, 0, stream>>>(Ctx, WtO, bo, out);
}
